// SuperPointBNNet_4647154614360
// MI455X (gfx1250) — matrix operand order repaired and checked by reading, compiled, not run on hardware
//
#include <hip/hip_runtime.h>

// ---------------------------------------------------------------------------
// SuperPointBNNet forward for MI455X (gfx1250, wave32, WMMA + TDM).
// Convs: implicit GEMM on v_wmma_f32_16x16x32_f16 with 2x2 register blocking
// per wave (4 WMMAs per K-chunk). Weight tiles DMA'd into LDS by the Tensor
// Data Mover (tensor_load_to_lds + s_wait_tensorcnt) when available; LDS
// tiles held in fragment-swizzled order so fragment loads are 2x ds_load_b128.
// BN scale/shift fused in the epilogue.
// ---------------------------------------------------------------------------

typedef _Float16 half_t;
typedef __attribute__((ext_vector_type(16))) _Float16     v16h;
typedef __attribute__((ext_vector_type(8)))  float        v8f;
typedef __attribute__((ext_vector_type(4)))  unsigned int v4u;
typedef __attribute__((ext_vector_type(8)))  int          v8i_t;
typedef __attribute__((ext_vector_type(4)))  int          v4i_t;

#if defined(__has_builtin)
#if __has_builtin(__builtin_amdgcn_tensor_load_to_lds) && \
    __has_builtin(__builtin_amdgcn_s_wait_tensorcnt)
#define USE_TDM 1
#endif
#endif
#ifndef USE_TDM
#define USE_TDM 0
#endif

#define DET_THRESH 0.001f
#define IOU_THRESH 0.1f
#define NMS_SIZE   4.0f
#define TOPK       300
#define NCAND      1024
#define NCAP       2048

#define TM 64    // output channels per block (4 x 16 tiles)
#define TN 128   // output pixels per block  (8 x 16 tiles)
#define TK 32    // K chunk (matches WMMA K)

// --------------------------- f32 -> f16 weights ----------------------------
__global__ void f32_to_f16_kernel(const float* __restrict__ in,
                                  half_t* __restrict__ out, int n) {
  int i = blockIdx.x * blockDim.x + threadIdx.x;
  if (i < n) out[i] = (half_t)in[i];
}

// ------------------- implicit-GEMM conv + BN (+ReLU), WMMA -----------------
// D[oc, n] = sum_k W[oc,k] * Im2col[k, n];  k = ci*ks*ks + ky*ks + kx
// n = ((b*Hy)+y)*Wx + x, SAME padding, stride 1.
// 8 waves: wm in {0,1} picks 32-oc half, wn in {0..3} picks 32-px quarter.
// Each wave computes a 2x2 grid of 16x16 WMMA tiles.
__global__ __launch_bounds__(256) void conv_wmma_kernel(
    const void* __restrict__ inp, int inIsF32,
    const half_t* __restrict__ w16,
    const float* __restrict__ scale, const float* __restrict__ bias,
    void* __restrict__ outp, int outIsF32,
    int Bn, int Ci, int Co, int Hy, int Wx, int ks, int relu) {
  // Fragment-swizzled tiles: sA[mt][lane*16+slot], sB[nt][lane*16+slot].
  __shared__ alignas(32) half_t sA[TM / 16][32 * 16];  // 4 KB
  __shared__ alignas(32) half_t sB[TN / 16][32 * 16];  // 8 KB
#if USE_TDM
  __shared__ alignas(32) half_t sRawA[TM * TK];        // 4 KB TDM landing pad
#endif

  const int tid  = threadIdx.x;
  const int lane = tid & 31;
  const int wave = tid >> 5;       // 0..7
  const int wm   = wave >> 2;      // 0..1 : oc half
  const int wn   = wave & 3;       // 0..3 : pixel quarter

  const int ocBlock = blockIdx.y * TM;
  const int nBlock  = (int)blockIdx.x * TN;
  const int HW   = Hy * Wx;
  const int Ntot = Bn * HW;
  const int KC   = Ci * ks * ks;
  const int pad  = ks >> 1;

  v8f acc00 = {}, acc01 = {}, acc10 = {}, acc11 = {};

#if USE_TDM
  const int waveUniform = __builtin_amdgcn_readfirstlane(wave);
  const unsigned int ldsRawAddr = (unsigned int)(size_t)&sRawA[0];
#endif

  for (int kc = 0; kc < KC; kc += TK) {
#if USE_TDM
    // ---- stage A (weights) via Tensor Data Mover: 2D tile [64 rows x 32 k]
    // D# per CDNA5 ISA (groups 0/1; groups 2/3 zero = 2D).
    // tensor_dim = remaining extent so TDM OOB zero-fill handles tile edges.
    if (waveUniform == 0) {
      unsigned long long ga =
          (unsigned long long)(size_t)(w16 + (size_t)ocBlock * KC + kc);
      int td0 = KC - kc;        // remaining K (elements)
      int td1 = Co - ocBlock;   // remaining rows
      v4u g0;
      g0.x = 1u;                                    // count=1, user mode
      g0.y = ldsRawAddr;                            // lds_addr (bytes)
      g0.z = (unsigned int)(ga & 0xFFFFFFFFull);    // global_addr[31:0]
      g0.w = (unsigned int)((ga >> 32) & 0x01FFFFFFull) | (2u << 30); // type=2
      v8i_t g1;
      g1[0] = 0x00010000;                           // data_size = 2 bytes
      g1[1] = (td0 & 0xFFFF) << 16;                 // tensor_dim0[15:0]
      g1[2] = ((td0 >> 16) & 0xFFFF) | ((td1 & 0xFFFF) << 16); // dim0 hi|dim1 lo
      g1[3] = ((td1 >> 16) & 0xFFFF) | (TK << 16);  // dim1 hi | tile_dim0
      g1[4] = TM;                                   // tile_dim1, tile_dim2=0
      g1[5] = KC;                                   // tensor_dim0_stride[31:0]
      g1[6] = 0;                                    // stride hi, dim1_stride lo
      g1[7] = 0;
      v4i_t gz4 = {0, 0, 0, 0};
      v8i_t gz8 = {0, 0, 0, 0, 0, 0, 0, 0};
      __builtin_amdgcn_tensor_load_to_lds(g0, g1, gz4, gz4, gz8, 0);
      __builtin_amdgcn_s_wait_tensorcnt(0);
    }
#else
    // ---- stage A (weights) into LDS, fragment-swizzled (direct path) ----
    for (int i = tid; i < TM * TK; i += 256) {
      int ml = i >> 5;         // 0..63
      int kl = i & 31;         // 0..31
      int oc = ocBlock + ml;
      int kg = kc + kl;
      half_t v = (half_t)0.f;
      if (oc < Co && kg < KC) v = w16[(size_t)oc * KC + kg];
      int mt    = ml >> 4;
      int laneA = (ml & 15) | (((kl >> 3) & 1) << 4);
      int slotA = ((kl >> 4) << 3) | (kl & 7);
      sA[mt][laneA * 16 + slotA] = v;
    }
    if (kc + TK < KC)
      __builtin_prefetch(&w16[(size_t)(ocBlock + (tid & 63)) * KC + kc + TK], 0, 1);
#endif

    // ---- stage B (im2col activations) into LDS, fragment-swizzled ----
    // i = kl*TN + nl so a wave reads 32 consecutive pixels at one k
    // (coalesced global gather).
    // B 32x16 f16 layout: element (k,n) -> lane (n&15)|(((k>>3)&1)<<4), slot ((k>>4)<<3)|(k&7)
    for (int i = tid; i < TK * TN; i += 256) {
      int kl = i >> 7;         // 0..31
      int nl = i & 127;        // 0..127
      int kg = kc + kl;
      int ng = nBlock + nl;
      half_t v = (half_t)0.f;
      if (kg < KC && ng < Ntot) {
        int b   = ng / HW;
        int rem = ng - b * HW;
        int y   = rem / Wx;
        int x   = rem - y * Wx;
        int ci, ky, kx;
        if (ks == 3) {
          ci = kg / 9;
          int r = kg - ci * 9;
          ky = r / 3;
          kx = r - ky * 3;
        } else {
          ci = kg; ky = 0; kx = 0;
        }
        int iy = y + ky - pad;
        int ix = x + kx - pad;
        if (iy >= 0 && iy < Hy && ix >= 0 && ix < Wx) {
          size_t off = (((size_t)b * Ci + ci) * Hy + iy) * Wx + ix;
          v = inIsF32 ? (half_t)((const float*)inp)[off]
                      : ((const half_t*)inp)[off];
        }
      }
      int nt    = nl >> 4;
      int laneB = (nl & 15) | (((kl >> 3) & 1) << 4);
      int slotB = ((kl >> 4) << 3) | (kl & 7);
      sB[nt][laneB * 16 + slotB] = v;
    }
    __syncthreads();

#if USE_TDM
    // ---- LDS->LDS swizzle of the TDM-landed row-major A tile into
    //      WMMA fragment order: (m,k) -> lane (m&15)|((k>>3&1)<<4),
    //                                   slot ((k>>4)<<3)|(k&7)
    for (int i = tid; i < TM * TK; i += 256) {
      int ml = i >> 5;
      int kl = i & 31;
      half_t v = sRawA[ml * TK + kl];
      int mt    = ml >> 4;
      int laneA = (ml & 15) | (((kl >> 3) & 1) << 4);
      int slotA = ((kl >> 4) << 3) | (kl & 7);
      sA[mt][laneA * 16 + slotA] = v;
    }
    __syncthreads();
#endif

    // ---- wide fragment loads (2x ds_load_b128 each) ----
    const v16h af0 = *(const v16h*)&sA[2 * wm + 0][lane * 16];
    const v16h af1 = *(const v16h*)&sA[2 * wm + 1][lane * 16];
    const v16h bf0 = *(const v16h*)&sB[2 * wn + 0][lane * 16];
    const v16h bf1 = *(const v16h*)&sB[2 * wn + 1][lane * 16];

    // ---- 2x2 register-blocked WMMA ----
    acc00 = __builtin_amdgcn_wmma_f32_16x16x32_f16(false, af0, false, bf0,
                                                   (short)0, acc00, false, false);
    acc01 = __builtin_amdgcn_wmma_f32_16x16x32_f16(false, af0, false, bf1,
                                                   (short)0, acc01, false, false);
    acc10 = __builtin_amdgcn_wmma_f32_16x16x32_f16(false, af1, false, bf0,
                                                   (short)0, acc10, false, false);
    acc11 = __builtin_amdgcn_wmma_f32_16x16x32_f16(false, af1, false, bf1,
                                                   (short)0, acc11, false, false);
    __syncthreads();
  }

  // ---- epilogue: C/D layout -> fused BN (+ReLU) -> store ----
  // C/D: lane&15 = column n, VGPR v -> row v + (lane<16 ? 0 : 8)
  const int laneN = lane & 15;
  const int laneM = (lane < 16) ? 0 : 8;
#pragma unroll
  for (int mi = 0; mi < 2; ++mi) {
#pragma unroll
    for (int ni = 0; ni < 2; ++ni) {
      const v8f* accp = (mi == 0) ? ((ni == 0) ? &acc00 : &acc01)
                                  : ((ni == 0) ? &acc10 : &acc11);
      int n = nBlock + (2 * wn + ni) * 16 + laneN;
      if (n < Ntot) {
        int b    = n / HW;
        int rem  = n - b * HW;
        int mOff = ocBlock + (2 * wm + mi) * 16 + laneM;
#pragma unroll
        for (int v = 0; v < 8; ++v) {
          int oc = mOff + v;
          if (oc < Co) {
            float r = (*accp)[v] * scale[oc] + bias[oc];
            if (relu) r = fmaxf(r, 0.f);
            size_t off = ((size_t)b * Co + oc) * HW + rem;
            if (outIsF32) ((float*)outp)[off] = r;
            else          ((half_t*)outp)[off] = (half_t)r;
          }
        }
      }
    }
  }
}

// ------------------------------- maxpool 2x2 -------------------------------
__global__ void maxpool_kernel(const half_t* __restrict__ in,
                               half_t* __restrict__ out,
                               int BC, int Ho, int Wo) {
  int i = blockIdx.x * blockDim.x + threadIdx.x;
  int total = BC * Ho * Wo;
  if (i >= total) return;
  int bc  = i / (Ho * Wo);
  int rem = i - bc * (Ho * Wo);
  int y   = rem / Wo;
  int x   = rem - y * Wo;
  int Wi  = Wo * 2;
  const half_t* p = in + (((size_t)bc * (Ho * 2)) + y * 2) * Wi + x * 2;
  float a = (float)p[0], b = (float)p[1], c = (float)p[Wi], d = (float)p[Wi + 1];
  out[i] = (half_t)fmaxf(fmaxf(a, b), fmaxf(c, d));
}

// -------------------- softmax over 65ch + pixel shuffle --------------------
__global__ void softmax_shuffle_kernel(const float* __restrict__ logits,
                                       float* __restrict__ prob,
                                       int Bn, int Hc, int Wc) {
  int i = blockIdx.x * blockDim.x + threadIdx.x;
  int total = Bn * Hc * Wc;
  if (i >= total) return;
  int b   = i / (Hc * Wc);
  int rem = i - b * (Hc * Wc);
  int yc  = rem / Wc;
  int xc  = rem - yc * Wc;
  int stride = Hc * Wc;
  const float* L = logits + (size_t)b * 65 * stride + yc * Wc + xc;

  float mx = -1e30f;
  for (int c = 0; c < 65; ++c) mx = fmaxf(mx, L[c * stride]);
  float sum = 0.f;
  for (int c = 0; c < 65; ++c) sum += __expf(L[c * stride] - mx);
  float inv = 1.f / sum;

  int Hp = Hc * 8, Wp = Wc * 8;
  for (int c = 0; c < 64; ++c) {
    int g1 = c >> 3, g2 = c & 7;
    float p = __expf(L[c * stride] - mx) * inv;
    prob[(size_t)b * Hp * Wp + (size_t)(yc * 8 + g1) * Wp + (xc * 8 + g2)] = p;
  }
}

// ------------- per-image top-1024 radix-select + bitonic + NMS -------------
__global__ __launch_bounds__(1024) void nms_kernel(
    const float* __restrict__ prob, float* __restrict__ prob_nms,
    float* __restrict__ pred_map, int Hn, int Wn) {
  __shared__ unsigned int hist[4096];
  __shared__ float sc[NCAP];
  __shared__ int   si[NCAP];
  __shared__ unsigned char keep[NCAND];
  __shared__ int sInt[4];

  const int tid = threadIdx.x;
  const int img = blockIdx.x;
  const int HW  = Hn * Wn;
  const float* P  = prob     + (size_t)img * HW;
  float*       PN = prob_nms + (size_t)img * HW;
  float*       PM = pred_map + (size_t)img * HW;

  // zero outputs + histogram
  for (int p = tid; p < HW; p += 1024) { PN[p] = 0.f; PM[p] = 0.f; }
  for (int h = tid; h < 4096; h += 1024) hist[h] = 0u;
  __syncthreads();

  // pass 1: histogram of top 12 float bits (all scores positive)
  for (int p = tid; p < HW; p += 1024) {
    float s = P[p];
    if (s > DET_THRESH) atomicAdd(&hist[__float_as_uint(s) >> 20], 1u);
  }
  __syncthreads();
  if (tid == 0) {
    unsigned int acc = 0; int pb = 0; unsigned int above = 0;
    for (int b2 = 4095; b2 >= 0; --b2) {
      unsigned int nb = acc + hist[b2];
      if (nb >= NCAND || b2 == 0) { pb = b2; above = acc; break; }
      acc = nb;
    }
    sInt[0] = pb; sInt[1] = (int)above;
  }
  __syncthreads();
  int pb1 = sInt[0];
  unsigned int above = (unsigned int)sInt[1];
  for (int h = tid; h < 4096; h += 1024) hist[h] = 0u;
  __syncthreads();

  // pass 2: refine within the pivot bin using next 12 bits
  for (int p = tid; p < HW; p += 1024) {
    float s = P[p];
    if (s > DET_THRESH) {
      unsigned int bits = __float_as_uint(s);
      if ((int)(bits >> 20) == pb1) atomicAdd(&hist[(bits >> 8) & 0xFFFu], 1u);
    }
  }
  __syncthreads();
  if (tid == 0) {
    unsigned int acc = above; int pb = 0;
    for (int b2 = 4095; b2 >= 0; --b2) {
      acc += hist[b2];
      if (acc >= NCAND || b2 == 0) { pb = b2; break; }
    }
    sInt[2] = (int)(((unsigned)pb1 << 20) | ((unsigned)pb << 8));
    sInt[3] = 0;
  }
  __syncthreads();
  unsigned int pivotBits = (unsigned int)sInt[2];

  // compaction of candidates >= pivot
  for (int p = tid; p < HW; p += 1024) {
    float s = P[p];
    if (s > DET_THRESH && __float_as_uint(s) >= pivotBits) {
      int pos = atomicAdd(&sInt[3], 1);
      if (pos < NCAP) { sc[pos] = s; si[pos] = p; }
    }
  }
  __syncthreads();
  int cnt = sInt[3]; if (cnt > NCAP) cnt = NCAP;
  for (int i = tid; i < NCAP; i += 1024)
    if (i >= cnt) { sc[i] = -1.f; si[i] = 0; }
  __syncthreads();

  // bitonic sort (descending by score), 2048 elements, 1024 threads
  for (int k = 2; k <= NCAP; k <<= 1) {
    for (int j = k >> 1; j > 0; j >>= 1) {
      for (int i = tid; i < NCAP; i += 1024) {
        int ixj = i ^ j;
        if (ixj > i) {
          bool desc = ((i & k) == 0);
          float a = sc[i], b2 = sc[ixj];
          bool sw = desc ? (a < b2) : (a > b2);
          if (sw) {
            sc[i] = b2; sc[ixj] = a;
            int t = si[i]; si[i] = si[ixj]; si[ixj] = t;
          }
        }
      }
      __syncthreads();
    }
  }

  // greedy NMS over top NCAND (boxes all 4x4 -> IOU from |dy|,|dx| only)
  for (int i = 0; i < NCAND; ++i) {
    if (tid == 0) sInt[3] = 0;
    __syncthreads();
    if (tid < i && keep[tid]) {
      int pi = si[i], pj = si[tid];
      float dy = fabsf((float)(pi / Wn) - (float)(pj / Wn));
      float dx = fabsf((float)(pi % Wn) - (float)(pj % Wn));
      float inter = fmaxf(0.f, NMS_SIZE - dy) * fmaxf(0.f, NMS_SIZE - dx);
      float iou = inter / (2.f * NMS_SIZE * NMS_SIZE - inter);
      if (iou > IOU_THRESH) sInt[3] = 1;
    }
    __syncthreads();
    if (tid == 0) keep[i] = (sc[i] > DET_THRESH) && (sInt[3] == 0);
    __syncthreads();
  }
  if (tid == 0) {
    int c2 = 0;
    for (int i = 0; i < NCAND; ++i)
      if (keep[i]) { ++c2; if (c2 > TOPK) keep[i] = 0; }
  }
  __syncthreads();
  if (tid < NCAND && keep[tid]) {
    PN[si[tid]] = sc[tid];
    PM[si[tid]] = sc[tid];   // pred_map == prob_nms here (kept > thresh)
  }
}

// --------------- bilinear x8 upsample + per-pixel L2 normalize -------------
// one block per output pixel; thread = channel; LDS tree reduction for norm
__global__ __launch_bounds__(256) void bilinear_l2_kernel(
    const float* __restrict__ raw, float* __restrict__ desc,
    int Bn, int C, int Hc, int Wc, int Ho, int Wo) {
  __shared__ float red[256];
  int p   = blockIdx.x;
  int b   = p / (Ho * Wo);
  int rem = p - b * (Ho * Wo);
  int y   = rem / Wo;
  int x   = rem - y * Wo;

  float sy = (y + 0.5f) * ((float)Hc / (float)Ho) - 0.5f;
  float sx = (x + 0.5f) * ((float)Wc / (float)Wo) - 0.5f;
  int y0 = (int)floorf(sy), x0 = (int)floorf(sx);
  float fy = sy - (float)y0, fx = sx - (float)x0;
  int y1 = min(max(y0 + 1, 0), Hc - 1);
  int x1 = min(max(x0 + 1, 0), Wc - 1);
  y0 = min(max(y0, 0), Hc - 1);
  x0 = min(max(x0, 0), Wc - 1);

  int c = threadIdx.x;
  const float* R = raw + ((size_t)b * C + c) * (size_t)Hc * Wc;
  float v00 = R[y0 * Wc + x0], v01 = R[y0 * Wc + x1];
  float v10 = R[y1 * Wc + x0], v11 = R[y1 * Wc + x1];
  float v = v00 * (1.f - fy) * (1.f - fx) + v01 * (1.f - fy) * fx +
            v10 * fy * (1.f - fx) + v11 * fy * fx;

  red[c] = v * v;
  __syncthreads();
  for (int s = 128; s > 0; s >>= 1) {
    if (c < s) red[c] += red[c + s];
    __syncthreads();
  }
  float nrm = sqrtf(red[0]) + 1e-12f;
  desc[((size_t)b * C + c) * (size_t)Ho * Wo + (size_t)y * Wo + x] = v / nrm;
}

// ------------------------------ host launcher ------------------------------
extern "C" void kernel_launch(void* const* d_in, const int* in_sizes, int n_in,
                              void* d_out, int out_size, void* d_ws,
                              size_t ws_size, hipStream_t stream) {
  (void)in_sizes; (void)n_in; (void)out_size; (void)ws_size;
  const int Bn = 4, Hh = 240, Ww = 320;

  struct ConvL { int ci, co, ks, win; };
  const ConvL convs[12] = {
      {1, 64, 3, 1},    {64, 64, 3, 4},   {64, 64, 3, 7},   {64, 64, 3, 10},
      {64, 128, 3, 13}, {128, 128, 3, 16},{128, 128, 3, 19},{128, 128, 3, 22},
      {128, 256, 3, 25},{256, 65, 1, 28}, {128, 256, 3, 31},{256, 256, 1, 34}};

  // workspace layout
  half_t* w16base = (half_t*)d_ws;
  size_t woff[12]; size_t wacc = 0;
  for (int i = 0; i < 12; ++i) {
    woff[i] = wacc;
    wacc += (size_t)convs[i].co * convs[i].ci * convs[i].ks * convs[i].ks;
  }
  size_t base = ((wacc * sizeof(half_t)) + 255) & ~(size_t)255;
  size_t actBytes = (((size_t)Bn * 64 * Hh * Ww * sizeof(half_t)) + 255) & ~(size_t)255;
  half_t* actA = (half_t*)((char*)d_ws + base);
  half_t* actB = (half_t*)((char*)d_ws + base + actBytes);
  half_t* tmpT = (half_t*)((char*)d_ws + base + 2 * actBytes);

  // convert all conv weights to f16
  for (int i = 0; i < 12; ++i) {
    int n = convs[i].co * convs[i].ci * convs[i].ks * convs[i].ks;
    f32_to_f16_kernel<<<(n + 255) / 256, 256, 0, stream>>>(
        (const float*)d_in[convs[i].win], w16base + woff[i], n);
  }

  auto conv = [&](const void* in, int inF32, int li, void* out, int outF32,
                  int Hy, int Wx, int relu) {
    int Ntot = Bn * Hy * Wx;
    dim3 g((Ntot + TN - 1) / TN, (convs[li].co + TM - 1) / TM);
    conv_wmma_kernel<<<g, 256, 0, stream>>>(
        in, inF32, w16base + woff[li],
        (const float*)d_in[convs[li].win + 1],
        (const float*)d_in[convs[li].win + 2],
        out, outF32, Bn, convs[li].ci, convs[li].co, Hy, Wx, convs[li].ks, relu);
  };
  auto pool = [&](const half_t* in, half_t* outp, int Cc, int Ho, int Wo) {
    int total = Bn * Cc * Ho * Wo;
    maxpool_kernel<<<(total + 255) / 256, 256, 0, stream>>>(in, outp, Bn * Cc, Ho, Wo);
  };

  // output slices (floats), in reference return order
  float* out      = (float*)d_out;
  float* prob     = out;                    // [4,240,320]
  float* prob_nms = out + 307200;           // [4,240,320]
  float* pred_map = out + 614400;           // [4,240,320]
  float* logits   = out + 921600;           // [4,65,30,40]
  float* desc_raw = out + 1233600;          // [4,256,30,40]
  float* desc     = out + 2462400;          // [4,256,240,320]

  // ---- backbone ----
  conv(d_in[0], 1, 0, actA, 0, 240, 320, 1);
  conv(actA,    0, 1, actB, 0, 240, 320, 1);
  pool(actB, actA, 64, 120, 160);
  conv(actA, 0, 2, actB, 0, 120, 160, 1);
  conv(actB, 0, 3, actA, 0, 120, 160, 1);
  pool(actA, actB, 64, 60, 80);
  conv(actB, 0, 4, actA, 0, 60, 80, 1);
  conv(actA, 0, 5, actB, 0, 60, 80, 1);
  pool(actB, actA, 128, 30, 40);
  conv(actA, 0, 6, actB, 0, 30, 40, 1);
  conv(actB, 0, 7, actA, 0, 30, 40, 1);    // feat (f16) in actA

  // ---- detector head ----
  conv(actA, 0, 8, tmpT, 0, 30, 40, 1);
  conv(tmpT, 0, 9, logits, 1, 30, 40, 0);
  {
    int total = Bn * 30 * 40;
    softmax_shuffle_kernel<<<(total + 255) / 256, 256, 0, stream>>>(
        logits, prob, Bn, 30, 40);
  }
  nms_kernel<<<Bn, 1024, 0, stream>>>(prob, prob_nms, pred_map, 240, 320);

  // ---- descriptor head ----
  conv(actA, 0, 10, tmpT, 0, 30, 40, 1);
  conv(tmpT, 0, 11, desc_raw, 1, 30, 40, 0);
  bilinear_l2_kernel<<<Bn * Hh * Ww, 256, 0, stream>>>(
      desc_raw, desc, Bn, 256, 30, 40, Hh, Ww);
}
